// DocPolarBERTEncoder_27410481283225
// MI455X (gfx1250) — matrix operand order repaired and checked by reading, not compiled, not run on hardware
//
#include <hip/hip_runtime.h>
#include <hip/hip_bf16.h>

// ---------------------------------------------------------------------------
// DocPolarBERT encoder for MI455X (gfx1250): bf16 WMMA GEMMs, f32 accumulate.
// Weight panels staged to LDS via gfx1250 async-to-LDS DMA (double-buffered,
// synchronized with s_wait_asynccnt + workgroup barriers).
// ---------------------------------------------------------------------------

typedef __bf16 bf16;
typedef __attribute__((ext_vector_type(8)))  bf16  v8bf;
typedef __attribute__((ext_vector_type(16))) bf16  v16bf;
typedef __attribute__((ext_vector_type(8)))  float v8f;

#define DEVINL __device__ __forceinline__

constexpr int S_  = 1024;
constexpr int D_  = 768;
constexpr int NH_ = 12;
constexpr int HD_ = 64;
constexpr int NL_ = 4;
constexpr int DFF_ = 3072;
constexpr float EPS_ = 1e-12f;

DEVINL bf16 f2bf(float f) {
  unsigned u = __builtin_bit_cast(unsigned, f);
  unsigned r = u + 0x7FFFu + ((u >> 16) & 1u);     // round-to-nearest-even
  unsigned short h = (unsigned short)(r >> 16);
  return __builtin_bit_cast(bf16, h);
}

// A fragment: 16(M) x 32(K) bf16, row-major source with leading dim ld.
// ISA layout: lanes 0-15 row M=lane, K = {0..7, 16..23}; lanes 16-31 K = {8..15, 24..31}.
DEVINL v16bf load_a_frag(const bf16* base, int ld) {
  const int lane = threadIdx.x & 31;
  const bf16* p = base + (lane & 15) * ld + ((lane >> 4) << 3);
  v8bf lo = *(const v8bf*)(p);
  v8bf hi = *(const v8bf*)(p + 16);
  return __builtin_shufflevector(lo, hi, 0,1,2,3,4,5,6,7,8,9,10,11,12,13,14,15);
}

// B fragment: 32(K) x 16(N), sourced from row-major W (N x K), i.e. B = W^T.
// Lane holds column n = lane&15, K = (lane>>4)*8 + {0..7, 16..23} (matches A fragment).
DEVINL v16bf load_b_frag(const bf16* base, int ld) {
  const int lane = threadIdx.x & 31;
  const bf16* p = base + (lane & 15) * ld + ((lane >> 4) << 3);
  v8bf lo = *(const v8bf*)(p);
  v8bf hi = *(const v8bf*)(p + 16);
  return __builtin_shufflevector(lo, hi, 0,1,2,3,4,5,6,7,8,9,10,11,12,13,14,15);
}

DEVINL v8f wmma_bf16(v16bf a, v16bf b, v8f c) {
  return __builtin_amdgcn_wmma_f32_16x16x32_bf16(false, a, false, b, (short)0, c,
                                                 false, false);
}

// gfx1250 async DMA: per-lane 16B global -> LDS, tracked on ASYNCcnt.
DEVINL void async_copy_b128(unsigned lds_byte_addr, const void* gptr) {
  asm volatile("global_load_async_to_lds_b128 %0, %1, off"
               :: "v"(lds_byte_addr), "v"(gptr)
               : "memory");
}
template <int N> DEVINL void wait_asynccnt() {
  asm volatile("s_wait_asynccnt %0" :: "i"(N) : "memory");
}

// ---------------------------------------------------------------------------
// Generic GEMM: C(M x Nout) = act( alpha * (A @ W^T + bias) + resid )
// A: (M x K) bf16 row-major lda; W: (Nout x K) bf16 row-major ldw.
// Block = 256 threads = 8 waves computing a 128x64 C tile; the shared 64x32
// W panel per k-step is staged into LDS by async-to-LDS DMA, double-buffered.
// Grid = (M/128)*(Nout/64) blocks.
// ---------------------------------------------------------------------------
__global__ __launch_bounds__(256)
void gemm_xwT_kernel(const bf16* __restrict__ A, int lda,
                     const bf16* __restrict__ W, int ldw,
                     const float* __restrict__ bias,
                     const float* __restrict__ resid,
                     float* __restrict__ Cf, bf16* __restrict__ Cb, int ldc,
                     int M, int Nout, int K, float alpha, int act) {
  __shared__ __align__(16) bf16 Bs[2][64 * 32];   // two 4 KB W panels

  const int ntilesN = Nout >> 6;
  const int nt = blockIdx.x % ntilesN;
  const int mb = blockIdx.x / ntilesN;            // 128-row block index
  const int w  = threadIdx.x >> 5;
  const int lane = threadIdx.x & 31;
  const int mt = mb * 8 + w;                      // this wave's 16-row tile

  const bf16* Ab = A + (size_t)mt * 16 * lda;
  const bf16* Wb = W + (size_t)nt * 64 * ldw;

  // Each thread DMAs one 16B chunk of the 64x32 panel (256 chunks total).
  const int chunk = threadIdx.x;
  const int brow  = chunk >> 2;                   // panel row 0..63
  const int bsub  = chunk & 3;                    // 8-elem sub-chunk in row
  const unsigned ldsDst0 = (unsigned)(size_t)(&Bs[0][chunk * 8]);
  const unsigned ldsDst1 = (unsigned)(size_t)(&Bs[1][chunk * 8]);
  const bf16* gsrc = Wb + (size_t)brow * ldw + bsub * 8;

  v8f c0 = {}, c1 = {}, c2 = {}, c3 = {};
  const int nk = K >> 5;

  async_copy_b128(ldsDst0, gsrc);                 // fill panel 0 (k = 0)
  for (int i = 0; i < nk; ++i) {
    const int buf = i & 1;
    if (i + 1 < nk) {
      async_copy_b128(buf ? ldsDst0 : ldsDst1, gsrc + ((i + 1) << 5));
      wait_asynccnt<1>();                         // oldest (this panel) done
    } else {
      wait_asynccnt<0>();
    }
    __syncthreads();                              // all 8 waves' slices visible

    v16bf a = load_a_frag(Ab + (i << 5), lda);
    const bf16* bb = &Bs[buf][0];
    c0 = wmma_bf16(a, load_b_frag(bb,           32), c0);
    c1 = wmma_bf16(a, load_b_frag(bb + 16 * 32, 32), c1);
    c2 = wmma_bf16(a, load_b_frag(bb + 32 * 32, 32), c2);
    c3 = wmma_bf16(a, load_b_frag(bb + 48 * 32, 32), c3);

    __syncthreads();                              // safe to overwrite this panel
  }

  const int colb = nt * 64 + (lane & 15);
  const int row0 = mt * 16 + ((lane >> 4) << 3);
  v8f accs[4] = {c0, c1, c2, c3};
#pragma unroll
  for (int s = 0; s < 4; ++s) {
    const int col = colb + s * 16;
    const float bv = bias ? bias[col] : 0.0f;
#pragma unroll
    for (int r = 0; r < 8; ++r) {
      const size_t off = (size_t)(row0 + r) * ldc + col;
      float v = (accs[s][r] + bv) * alpha;
      if (resid) v += resid[off];
      if (act) v = 0.5f * v * (1.0f + erff(v * 0.70710678118654752f));  // exact GELU
      if (Cf) Cf[off] = v;
      if (Cb) Cb[off] = f2bf(v);
    }
  }
}

// ---------------------------------------------------------------------------
// Rel-position score add: for each query position s,
//   scores[:, s, :] += Q_row(s) viewed as (12 x 64) @ rel_sum[s]^T (64 x 1024)
// One workgroup per s; A (padded 16x64) staged in LDS; 8 waves x 8 col-tiles.
// ---------------------------------------------------------------------------
__global__ __launch_bounds__(256)
void scores_rel_kernel(const bf16* __restrict__ Qb,   // (S, D) bf16 (pre-scaled q)
                       const bf16* __restrict__ rel,  // (S, S, HD) bf16
                       float* __restrict__ scores) {  // (NH, S, S) f32
  const int s = blockIdx.x;
  __shared__ __align__(16) bf16 As[16 * 64];
  const int t = threadIdx.x;
#pragma unroll
  for (int i = 0; i < 4; ++i) {
    const int idx = t * 4 + i;                       // 0..1023 over 16x64
    As[idx] = (idx < NH_ * HD_) ? Qb[(size_t)s * D_ + idx] : f2bf(0.0f);
  }
  __syncthreads();

  const int w = t >> 5;
  const int lane = t & 31;
  const v16bf a0 = load_a_frag(As, 64);
  const v16bf a1 = load_a_frag(As + 32, 64);
  const int col = lane & 15;
  const int r0 = (lane >> 4) << 3;

  for (int j = 0; j < 8; ++j) {
    const int ct = w * 8 + j;                        // key-column tile 0..63
    const bf16* Wb = rel + ((size_t)s * S_ + (size_t)ct * 16) * HD_;
    if (j < 7)                                       // hint the streaming rel fetch
      __builtin_prefetch(Wb + 16 * HD_, 0, 1);
    const v16bf b0 = load_b_frag(Wb, HD_);
    const v16bf b1 = load_b_frag(Wb + 32, HD_);
    v8f c = {};
#pragma unroll
    for (int r = 0; r < 8; ++r) {
      const int head = r0 + r;
      c[r] = (head < NH_)
                 ? scores[((size_t)head * S_ + s) * S_ + ct * 16 + col]
                 : 0.0f;
    }
    c = wmma_bf16(a0, b0, c);
    c = wmma_bf16(a1, b1, c);
#pragma unroll
    for (int r = 0; r < 8; ++r) {
      const int head = r0 + r;
      if (head < NH_)
        scores[((size_t)head * S_ + s) * S_ + ct * 16 + col] = c[r];
    }
  }
}

// ---------------------------------------------------------------------------
// Stable softmax over keys (PB-relax is shift-invariant => plain softmax+mask).
// ---------------------------------------------------------------------------
__global__ __launch_bounds__(256)
void softmax_kernel(const float* __restrict__ scores,
                    const float* __restrict__ mask,
                    bf16* __restrict__ probs) {
  const int row = blockIdx.x;                        // n*S + s
  const float* x = scores + (size_t)row * S_;
  bf16* p = probs + (size_t)row * S_;
  const int t = threadIdx.x;
  __shared__ float red[256];

  float vals[4];
  float m = -3.0e38f;
#pragma unroll
  for (int i = 0; i < 4; ++i) {
    const int c = t + i * 256;
    vals[i] = x[c] + mask[c];
    m = fmaxf(m, vals[i]);
  }
  red[t] = m; __syncthreads();
  for (int st = 128; st > 0; st >>= 1) {
    if (t < st) red[t] = fmaxf(red[t], red[t + st]);
    __syncthreads();
  }
  m = red[0]; __syncthreads();

  float sum = 0.0f;
#pragma unroll
  for (int i = 0; i < 4; ++i) { vals[i] = expf(vals[i] - m); sum += vals[i]; }
  red[t] = sum; __syncthreads();
  for (int st = 128; st > 0; st >>= 1) {
    if (t < st) red[t] += red[t + st];
    __syncthreads();
  }
  const float inv = 1.0f / red[0];
#pragma unroll
  for (int i = 0; i < 4; ++i) p[t + i * 256] = f2bf(vals[i] * inv);
}

// ---------------------------------------------------------------------------
// Row LayerNorm over D=768 (3 elems/thread), dual f32 + bf16 output.
// ---------------------------------------------------------------------------
__global__ __launch_bounds__(256)
void layernorm_kernel(const float* __restrict__ x, const float* __restrict__ g,
                      const float* __restrict__ b, float* __restrict__ yf,
                      bf16* __restrict__ yb) {
  const int row = blockIdx.x;
  const float* xr = x + (size_t)row * D_;
  const int t = threadIdx.x;
  __shared__ float red[256];

  const float v0 = xr[t], v1 = xr[t + 256], v2 = xr[t + 512];
  red[t] = v0 + v1 + v2; __syncthreads();
  for (int st = 128; st > 0; st >>= 1) { if (t < st) red[t] += red[t + st]; __syncthreads(); }
  const float mu = red[0] * (1.0f / D_); __syncthreads();

  const float d0 = v0 - mu, d1 = v1 - mu, d2 = v2 - mu;
  red[t] = d0 * d0 + d1 * d1 + d2 * d2; __syncthreads();
  for (int st = 128; st > 0; st >>= 1) { if (t < st) red[t] += red[t + st]; __syncthreads(); }
  const float rs = rsqrtf(red[0] * (1.0f / D_) + EPS_);

  const float o0 = d0 * rs * g[t]       + b[t];
  const float o1 = d1 * rs * g[t + 256] + b[t + 256];
  const float o2 = d2 * rs * g[t + 512] + b[t + 512];
  float* yr = yf + (size_t)row * D_;
  bf16* ybr = yb + (size_t)row * D_;
  yr[t] = o0; yr[t + 256] = o1; yr[t + 512] = o2;
  ybr[t] = f2bf(o0); ybr[t + 256] = f2bf(o1); ybr[t + 512] = f2bf(o2);
}

// --------------------------- small utility kernels -------------------------
__global__ void cvt_bf16_kernel(const float* __restrict__ src,
                                bf16* __restrict__ dst, int n) {
  int i = blockIdx.x * blockDim.x + threadIdx.x;
  const int stride = gridDim.x * blockDim.x;
  for (; i < n; i += stride) dst[i] = f2bf(src[i]);
}

__global__ void cvt_dual_kernel(const float* __restrict__ src,
                                float* __restrict__ dstf,
                                bf16* __restrict__ dstb, int n) {
  int i = blockIdx.x * blockDim.x + threadIdx.x;
  const int stride = gridDim.x * blockDim.x;
  for (; i < n; i += stride) {
    const float v = src[i];
    dstf[i] = v;
    dstb[i] = f2bf(v);
  }
}

__global__ void rel_sum_kernel(const float* __restrict__ a,
                               const float* __restrict__ b,
                               const float* __restrict__ c,
                               bf16* __restrict__ out, int n) {
  int i = blockIdx.x * blockDim.x + threadIdx.x;
  const int stride = gridDim.x * blockDim.x;
  for (; i < n; i += stride) out[i] = f2bf(a[i] + b[i] + c[i]);
}

// vT[n][h][k] = V[k][n*64+h] : makes PV GEMM's B operand row-major (HD x S).
__global__ void transpose_v_kernel(const bf16* __restrict__ Vb,
                                   bf16* __restrict__ vT) {
  const int idx = blockIdx.x * blockDim.x + threadIdx.x;  // NH*HD*S = 786432
  const int n = idx >> 16;
  const int h = (idx >> 10) & 63;
  const int k = idx & 1023;
  vT[idx] = Vb[(size_t)k * D_ + n * 64 + h];
}

// ---------------------------------------------------------------------------
extern "C" void kernel_launch(void* const* d_in, const int* in_sizes, int n_in,
                              void* d_out, int out_size, void* d_ws, size_t ws_size,
                              hipStream_t stream) {
  (void)in_sizes; (void)n_in; (void)out_size; (void)ws_size;

  const float* hidden = (const float*)d_in[0];
  const float* mask   = (const float*)d_in[1];
  const float* rel0   = (const float*)d_in[2];
  const float* rel1   = (const float*)d_in[3];
  const float* rel2   = (const float*)d_in[4];
  const float* Wq  = (const float*)d_in[5];  const float* bq  = (const float*)d_in[6];
  const float* Wk  = (const float*)d_in[7];  const float* bk  = (const float*)d_in[8];
  const float* Wv  = (const float*)d_in[9];  const float* bv  = (const float*)d_in[10];
  const float* Wo  = (const float*)d_in[11]; const float* bo  = (const float*)d_in[12];
  const float* ln1g = (const float*)d_in[13]; const float* ln1b = (const float*)d_in[14];
  const float* Wi  = (const float*)d_in[15]; const float* bi  = (const float*)d_in[16];
  const float* Wo2 = (const float*)d_in[17]; const float* bo2 = (const float*)d_in[18];
  const float* ln2g = (const float*)d_in[19]; const float* ln2b = (const float*)d_in[20];

  // ---- workspace carve ----
  char* p = (char*)d_ws;
  auto alloc = [&](size_t bytes) -> void* {
    void* r = (void*)p;
    p += (bytes + 255) & ~(size_t)255;
    return r;
  };
  bf16* wq_b  = (bf16*)alloc((size_t)NL_ * D_ * D_ * 2);
  bf16* wk_b  = (bf16*)alloc((size_t)NL_ * D_ * D_ * 2);
  bf16* wv_b  = (bf16*)alloc((size_t)NL_ * D_ * D_ * 2);
  bf16* wo_b  = (bf16*)alloc((size_t)NL_ * D_ * D_ * 2);
  bf16* wi_b  = (bf16*)alloc((size_t)NL_ * DFF_ * D_ * 2);
  bf16* wo2_b = (bf16*)alloc((size_t)NL_ * D_ * DFF_ * 2);
  bf16* relsum = (bf16*)alloc((size_t)S_ * S_ * HD_ * 2);          // 128 MB
  float* h_f   = (float*)alloc((size_t)S_ * D_ * 4);
  bf16*  h_b   = (bf16*)alloc((size_t)S_ * D_ * 2);
  bf16*  q_b   = (bf16*)alloc((size_t)S_ * D_ * 2);
  bf16*  k_b   = (bf16*)alloc((size_t)S_ * D_ * 2);
  bf16*  v_b   = (bf16*)alloc((size_t)S_ * D_ * 2);
  bf16*  vT_b  = (bf16*)alloc((size_t)NH_ * HD_ * S_ * 2);
  float* scores = (float*)alloc((size_t)NH_ * S_ * S_ * 4);        // 48 MB
  bf16*  probs  = (bf16*)alloc((size_t)NH_ * S_ * S_ * 2);         // 24 MB
  bf16*  ctx_b  = (bf16*)alloc((size_t)S_ * D_ * 2);
  float* tmp_f  = (float*)alloc((size_t)S_ * D_ * 4);
  float* attn_f = (float*)alloc((size_t)S_ * D_ * 4);
  bf16*  attn_b = (bf16*)alloc((size_t)S_ * D_ * 2);
  bf16*  inter_b = (bf16*)alloc((size_t)S_ * DFF_ * 2);

  const int CT = 256;

  // ---- one-time conversions ----
  cvt_bf16_kernel<<<1024, CT, 0, stream>>>(Wq,  wq_b,  NL_ * D_ * D_);
  cvt_bf16_kernel<<<1024, CT, 0, stream>>>(Wk,  wk_b,  NL_ * D_ * D_);
  cvt_bf16_kernel<<<1024, CT, 0, stream>>>(Wv,  wv_b,  NL_ * D_ * D_);
  cvt_bf16_kernel<<<1024, CT, 0, stream>>>(Wo,  wo_b,  NL_ * D_ * D_);
  cvt_bf16_kernel<<<2048, CT, 0, stream>>>(Wi,  wi_b,  NL_ * DFF_ * D_);
  cvt_bf16_kernel<<<2048, CT, 0, stream>>>(Wo2, wo2_b, NL_ * D_ * DFF_);
  rel_sum_kernel<<<8192, CT, 0, stream>>>(rel0, rel1, rel2, relsum, S_ * S_ * HD_);
  cvt_dual_kernel<<<768, CT, 0, stream>>>(hidden, h_f, h_b, S_ * D_);

  for (int l = 0; l < NL_; ++l) {
    const bf16* wq_l  = wq_b  + (size_t)l * D_ * D_;
    const bf16* wk_l  = wk_b  + (size_t)l * D_ * D_;
    const bf16* wv_l  = wv_b  + (size_t)l * D_ * D_;
    const bf16* wo_l  = wo_b  + (size_t)l * D_ * D_;
    const bf16* wi_l  = wi_b  + (size_t)l * DFF_ * D_;
    const bf16* wo2_l = wo2_b + (size_t)l * D_ * DFF_;

    // Q (pre-scaled by 1/sqrt(HD)), K, V projections: (1024x768)@(768x768)^T
    // grid = (M/128)*(Nout/64)
    gemm_xwT_kernel<<<96, CT, 0, stream>>>(h_b, D_, wq_l, D_, bq + l * D_, nullptr,
                                           nullptr, q_b, D_, S_, D_, D_, 0.125f, 0);
    gemm_xwT_kernel<<<96, CT, 0, stream>>>(h_b, D_, wk_l, D_, bk + l * D_, nullptr,
                                           nullptr, k_b, D_, S_, D_, D_, 1.0f, 0);
    gemm_xwT_kernel<<<96, CT, 0, stream>>>(h_b, D_, wv_l, D_, bv + l * D_, nullptr,
                                           nullptr, v_b, D_, S_, D_, D_, 1.0f, 0);

    // scores[n] = q_n @ k_n^T  (per head: 1024x1024, K=64)
    for (int n = 0; n < NH_; ++n)
      gemm_xwT_kernel<<<128, CT, 0, stream>>>(q_b + n * HD_, D_, k_b + n * HD_, D_,
                                              nullptr, nullptr,
                                              scores + (size_t)n * S_ * S_, nullptr,
                                              S_, S_, S_, HD_, 1.0f, 0);

    // scores[:, s, :] += q_heads(s) @ rel_sum[s]^T
    scores_rel_kernel<<<S_, CT, 0, stream>>>(q_b, relsum, scores);

    // softmax (+mask) -> bf16 probs
    softmax_kernel<<<NH_ * S_, CT, 0, stream>>>(scores, mask, probs);

    // ctx[n] = probs_n @ v_n  (needs v^T row-major)
    transpose_v_kernel<<<(NH_ * HD_ * S_) / CT, CT, 0, stream>>>(v_b, vT_b);
    for (int n = 0; n < NH_; ++n)
      gemm_xwT_kernel<<<8, CT, 0, stream>>>(probs + (size_t)n * S_ * S_, S_,
                                            vT_b + (size_t)n * HD_ * S_, S_,
                                            nullptr, nullptr, nullptr,
                                            ctx_b + n * HD_, D_,
                                            S_, HD_, S_, 1.0f, 0);

    // attn_out = LN1(ctx @ Wo^T + bo + h)
    gemm_xwT_kernel<<<96, CT, 0, stream>>>(ctx_b, D_, wo_l, D_, bo + l * D_, h_f,
                                           tmp_f, nullptr, D_, S_, D_, D_, 1.0f, 0);
    layernorm_kernel<<<S_, CT, 0, stream>>>(tmp_f, ln1g + l * D_, ln1b + l * D_,
                                            attn_f, attn_b);

    // inter = gelu(attn_out @ Wi^T + bi)   (1024x3072)
    gemm_xwT_kernel<<<384, CT, 0, stream>>>(attn_b, D_, wi_l, D_, bi + l * DFF_,
                                            nullptr, nullptr, inter_b, DFF_,
                                            S_, DFF_, D_, 1.0f, 1);

    // h = LN2(inter @ Wo2^T + bo2 + attn_out)
    gemm_xwT_kernel<<<96, CT, 0, stream>>>(inter_b, DFF_, wo2_l, DFF_, bo2 + l * D_,
                                           attn_f, tmp_f, nullptr, D_,
                                           S_, D_, DFF_, 1.0f, 0);
    float* hout = (l == NL_ - 1) ? (float*)d_out : h_f;
    layernorm_kernel<<<S_, CT, 0, stream>>>(tmp_f, ln2g + l * D_, ln2b + l * D_,
                                            hout, h_b);
  }
}
